// LearnablePixelwiseAnisoJBU_NoParent_24610162606566
// MI455X (gfx1250) — matrix operand order repaired and checked by reading, not compiled, not run on hardware
//
#include <hip/hip_runtime.h>
#include <hip/hip_bf16.h>

typedef __attribute__((ext_vector_type(16))) _Float16 v16h;
typedef __attribute__((ext_vector_type(8)))  _Float16 v8h;
typedef __attribute__((ext_vector_type(8)))  float    v8f;

#define HL 16
#define WL 16
#define CFEAT 384
#define HH 256
#define WH 256
#define RMAX 8
#define NP 256          // LR pixels = HL*WL
#define NEGBIG (-1e30f)

__device__ __forceinline__ int iclamp(int v, int lo, int hi) {
    return v < lo ? lo : (v > hi ? hi : v);
}

// ---------------------------------------------------------------------------
// Prep kernel: per-LR-pixel parameter tables + f16 feature matrix.
//   params[p][8] = { cos(th), sin(th), 1/(2sx^2+1e-8), 1/(2sy^2+1e-8),
//                    1/(2sr^2+1e-8), guide_lr_r, guide_lr_g, guide_lr_b }
//   maxsig[p]    = max(sx, sy)   (pre-clamp, for R_map bilinear)
//   feat16[c][p] = (f16) feat[c][p]
// ---------------------------------------------------------------------------
__global__ __launch_bounds__(256) void jbu_prep(
    const float* __restrict__ feat,
    const float* __restrict__ guide,
    const float* __restrict__ sx_raw,
    const float* __restrict__ sy_raw,
    const float* __restrict__ th_raw,
    const float* __restrict__ sr_raw,
    float* __restrict__ params,
    float* __restrict__ maxsig,
    _Float16* __restrict__ feat16)
{
    if (blockIdx.x == 0) {
        const int p = threadIdx.x;          // LR pixel
        const int i = p >> 4, j = p & 15;
        const float sx = __expf(sx_raw[p]);
        const float sy = __expf(sy_raw[p]);
        const float th = 3.14159265358979f * tanhf(th_raw[p]);
        const float sr = __expf(sr_raw[p]);
        maxsig[p] = fmaxf(sx, sy);
        const float sxm = fmaxf(sx, 1e-6f);
        const float sym = fmaxf(sy, 1e-6f);
        const float srm = fmaxf(sr, 1e-6f);
        float* pp = params + p * 8;
        pp[0] = __cosf(th);
        pp[1] = __sinf(th);
        pp[2] = 1.0f / (2.0f * sxm * sxm + 1e-8f);
        pp[3] = 1.0f / (2.0f * sym * sym + 1e-8f);
        pp[4] = 1.0f / (2.0f * srm * srm + 1e-8f);
        // bilinear downsample of guide 256->16: sample sits exactly between
        // rows/cols 16k+7 and 16k+8 (weights 0.5/0.5) -> 2x2 average.
        const int y0 = 16 * i + 7, y1 = 16 * i + 8;
        const int x0 = 16 * j + 7, x1 = 16 * j + 8;
        #pragma unroll
        for (int c = 0; c < 3; ++c) {
            const float* g = guide + c * (HH * WH);
            pp[5 + c] = 0.25f * (g[y0 * WH + x0] + g[y0 * WH + x1] +
                                 g[y1 * WH + x0] + g[y1 * WH + x1]);
        }
    }
    // f32 -> f16 feature conversion (grid-stride)
    for (int idx = blockIdx.x * blockDim.x + threadIdx.x; idx < CFEAT * NP;
         idx += gridDim.x * blockDim.x)
        feat16[idx] = (_Float16)feat[idx];
}

// ---------------------------------------------------------------------------
// Main fused kernel: one block (256 thr = 8 waves) per HR row.
//   Stage A: per-thread two-pass softmax over 17x17 neighbors into LDS
//            acc[p][t] (256x256 f32) + den[t].
//   Stage B: 384x256 = feat16[384,256] x acc[256,256] via v_wmma, /den,
//            store to out. mt unrolled x2 -> two independent WMMA chains.
// Dynamic LDS: acc 256KB | den 1KB | params 8KB = 271360 B (< 320KB WGP LDS).
// ---------------------------------------------------------------------------
__global__ __launch_bounds__(256) void jbu_main(
    const float* __restrict__ guide,
    const float* __restrict__ params_g,
    const float* __restrict__ maxsig,
    const _Float16* __restrict__ feat16,
    float* __restrict__ out)
{
    extern __shared__ float smem[];
    float* acc  = smem;                  // [256 p][256 t]
    float* denl = smem + NP * 256;       // [256]
    float* prm  = denl + 256;            // [256][8]

    const int tid = threadIdx.x;
    const int h   = blockIdx.x;          // HR row
    const int w   = tid;                 // HR col
    const int q0  = h * WH;

    // stage param tables into LDS
    {
        const float* src = params_g + tid * 8;
        float* dst = prm + tid * 8;
        #pragma unroll
        for (int k = 0; k < 8; ++k) dst[k] = src[k];
    }
    // zero own acc column (bank-conflict-free: bank = tid & 63)
    for (int p = 0; p < NP; ++p) acc[p * 256 + tid] = 0.0f;
    __syncthreads();

    // ---- Stage A: weights -------------------------------------------------
    const float u = ((float)h - 7.5f) * 0.0625f;   // (h+0.5)/16 - 0.5
    const float v = ((float)w - 7.5f) * 0.0625f;
    const int uc = iclamp((int)rintf(u), 0, HL - 1);
    const int vc = iclamp((int)rintf(v), 0, WL - 1);

    // R_map = clip(ceil(2 * bilerp(max(sx,sy))), 1, 8)
    const float ys = fmaxf(u, 0.0f), xs = fmaxf(v, 0.0f);
    const int y0 = (int)floorf(ys), x0 = (int)floorf(xs);
    const int y1 = min(y0 + 1, HL - 1), x1 = min(x0 + 1, WL - 1);
    const float wy = ys - (float)y0, wx = xs - (float)x0;
    const float se =
        (1.0f - wy) * ((1.0f - wx) * maxsig[y0 * WL + x0] + wx * maxsig[y0 * WL + x1]) +
        wy          * ((1.0f - wx) * maxsig[y1 * WL + x0] + wx * maxsig[y1 * WL + x1]);
    const float R  = fminf(fmaxf(ceilf(2.0f * se), 1.0f), (float)RMAX);
    const float R2 = R * R;

    const float gr = guide[0 * HH * WH + q0 + w];
    const float gg = guide[1 * HH * WH + q0 + w];
    const float gb = guide[2 * HH * WH + q0 + w];

    // pass 1: running max of log-weights
    float mx = NEGBIG;
    for (int dY = -RMAX; dY <= RMAX; ++dY) {
        const int Ui = iclamp(uc + dY, 0, HL - 1);
        const float dyv = (float)h - (16.0f * (float)Ui + 7.5f);
        for (int dX = -RMAX; dX <= RMAX; ++dX) {
            if ((float)(dY * dY + dX * dX) > R2) continue;
            const int Vi  = iclamp(vc + dX, 0, WL - 1);
            const int pix = Ui * WL + Vi;
            const float* pp = prm + pix * 8;
            const float dx = (float)w - (16.0f * (float)Vi + 7.5f);
            const float ct = pp[0], st = pp[1];
            const float a =  dx * ct + dyv * st;
            const float b = -dx * st + dyv * ct;
            const float d0 = gr - pp[5], d1 = gg - pp[6], d2 = gb - pp[7];
            const float gd2 = d0 * d0 + d1 * d1 + d2 * d2;
            const float lw = -(a * a) * pp[2] - (b * b) * pp[3] - gd2 * pp[4];
            mx = fmaxf(mx, lw);
        }
    }
    // pass 2: exp-accumulate into LDS acc + denominator
    float den = 0.0f;
    for (int dY = -RMAX; dY <= RMAX; ++dY) {
        const int Ui = iclamp(uc + dY, 0, HL - 1);
        const float dyv = (float)h - (16.0f * (float)Ui + 7.5f);
        for (int dX = -RMAX; dX <= RMAX; ++dX) {
            if ((float)(dY * dY + dX * dX) > R2) continue;
            const int Vi  = iclamp(vc + dX, 0, WL - 1);
            const int pix = Ui * WL + Vi;
            const float* pp = prm + pix * 8;
            const float dx = (float)w - (16.0f * (float)Vi + 7.5f);
            const float ct = pp[0], st = pp[1];
            const float a =  dx * ct + dyv * st;
            const float b = -dx * st + dyv * ct;
            const float d0 = gr - pp[5], d1 = gg - pp[6], d2 = gb - pp[7];
            const float gd2 = d0 * d0 + d1 * d1 + d2 * d2;
            const float lw = -(a * a) * pp[2] - (b * b) * pp[3] - gd2 * pp[4];
            const float s = __expf(lw - mx);
            acc[pix * 256 + tid] += s;
            den += s;
        }
    }
    denl[tid] = den;
    __syncthreads();

    // ---- Stage B: WMMA GEMM ----------------------------------------------
    // out[c, q0+q] = sum_p feat16[c,p] * acc[p,q] / den[q]
    // 24 M-tiles x 16 N-tiles, K = 256 (8 steps of 32). 8 waves -> 2 N-tiles/wave.
    const int lane = tid & 31;
    const int wave = tid >> 5;
    const int krow = lane >> 4;    // 0 = low lane half, 1 = high
    const int nlan = lane & 15;

    for (int nt = wave * 2; nt < wave * 2 + 2; ++nt) {
        // B 32x16 f16 tiles for all 8 k-steps, cached in VGPRs.
        // Lane layout: column N = lane&15; K-half = lane>>4; elem e -> K = khalf*16+e.
        v16h Bt[8];
        const int col = nt * 16 + nlan;
        #pragma unroll
        for (int kk = 0; kk < 8; ++kk) {
            const int kb = kk * 32 + krow * 16;
            v16h bt;
            #pragma unroll
            for (int e = 0; e < 16; ++e)
                bt[e] = (_Float16)acc[(kb + e) * 256 + col];
            Bt[kk] = bt;
        }
        const float rd = 1.0f / fmaxf(denl[col], 1e-8f);

        // Two independent accumulation chains (mt unrolled x2) so the matrix
        // pipe can overlap one chain's A-tile loads with the other's WMMAs.
        for (int mt = 0; mt < 24; mt += 2) {
            // A 16x32 f16: lane row M = lane&15; k-base = (lane>=16)?16:0, elem e -> K = kb+e
            const _Float16* ap0 = feat16 + ((mt + 0) * 16 + nlan) * NP + krow * 16;
            const _Float16* ap1 = feat16 + ((mt + 1) * 16 + nlan) * NP + krow * 16;
            v8f c0 = {};
            v8f c1 = {};
            #pragma unroll
            for (int kk = 0; kk < 8; ++kk) {
                const v8h* pl0 = (const v8h*)(ap0 + kk * 32);
                const v8h* pl1 = (const v8h*)(ap1 + kk * 32);
                const v8h lo0 = pl0[0];      // K = kb .. kb+7
                const v8h hi0 = pl0[1];      // K = kb+8 .. kb+15
                const v8h lo1 = pl1[0];
                const v8h hi1 = pl1[1];
                v16h a0, a1;
                #pragma unroll
                for (int e = 0; e < 8; ++e) {
                    a0[e] = lo0[e]; a0[8 + e] = hi0[e];
                    a1[e] = lo1[e]; a1[8 + e] = hi1[e];
                }
                c0 = __builtin_amdgcn_wmma_f32_16x16x32_f16(
                    false, a0, false, Bt[kk], (short)0, c0, false, false);
                c1 = __builtin_amdgcn_wmma_f32_16x16x32_f16(
                    false, a1, false, Bt[kk], (short)0, c1, false, false);
            }
            // D 16x16 f32: VGPR r -> M = r + 8*krow, N = lane&15
            const int ch0 = (mt + 0) * 16 + krow * 8;
            const int ch1 = (mt + 1) * 16 + krow * 8;
            const int qq  = q0 + nt * 16 + nlan;
            #pragma unroll
            for (int r = 0; r < 8; ++r) {
                out[(ch0 + r) * (HH * WH) + qq] = c0[r] * rd;
                out[(ch1 + r) * (HH * WH) + qq] = c1[r] * rd;
            }
        }
    }
}

// ---------------------------------------------------------------------------
extern "C" void kernel_launch(void* const* d_in, const int* in_sizes, int n_in,
                              void* d_out, int out_size, void* d_ws, size_t ws_size,
                              hipStream_t stream) {
    (void)in_sizes; (void)n_in; (void)out_size; (void)ws_size;
    const float* feat   = (const float*)d_in[0];  // [1,384,16,16]
    const float* guide  = (const float*)d_in[1];  // [1,3,256,256]
    const float* sx_raw = (const float*)d_in[2];  // [1,1,16,16]
    const float* sy_raw = (const float*)d_in[3];
    const float* th_raw = (const float*)d_in[4];
    const float* sr_raw = (const float*)d_in[5];
    float* out = (float*)d_out;                    // [1,384,256,256]

    // workspace layout
    float*    params = (float*)d_ws;                                   // 256*8 f32
    float*    msig   = params + NP * 8;                                // 256 f32
    _Float16* feat16 = (_Float16*)((char*)d_ws + (NP * 8 + NP) * 4);   // 384*256 f16 (16B aligned)

    jbu_prep<<<384, 256, 0, stream>>>(feat, guide, sx_raw, sy_raw, th_raw, sr_raw,
                                      params, msig, feat16);

    const size_t ldsBytes = (size_t)(NP * 256 + 256 + NP * 8) * sizeof(float); // 271360
    jbu_main<<<HH, 256, ldsBytes, stream>>>(guide, params, msig, feat16, out);
}
